// FlashNeoxAttention_2044404433538
// MI455X (gfx1250) — matrix operand order repaired and checked by reading, not compiled, not run on hardware
//
#include <hip/hip_runtime.h>
#include <hip/hip_bf16.h>
#include <stdint.h>

// ---------------------------------------------------------------------------
// FlashNeoxAttention for MI455X (gfx1250, wave32, WMMA).
// Pipeline: f32->bf16 convert / weight transpose -> bf16 WMMA QKV GEMM (f32 acc,
//           ping-pong double buffer + global_prefetch) -> RoPE + split (Q,K
//           row-major bf16; V transposed [h][d][t] bf16) -> per-wave flash
//           attention with v_wmma_f32_16x16x32_bf16 -> bf16 WMMA output GEMM.
// ---------------------------------------------------------------------------

typedef __attribute__((ext_vector_type(8)))  __bf16 v8bf;
typedef __attribute__((ext_vector_type(16))) __bf16 v16bf;
typedef __attribute__((ext_vector_type(8)))  float  v8f;

#define T_TOTAL 2048
#define NHEADS  16
#define HSIZE   128
#define HID     2048

// Load a 16-element bf16 fragment from two 16-byte-aligned 8-element chunks.
// A-operand layout (16x32, MxK): lane holds 8 contiguous K at (g*8) and 8 at (16+g*8).
// Contiguous order (PV step: A = Vt, B = P^T): lane holds 16 contiguous K at (g*16) -> p1 = p0+8.
__device__ __forceinline__ v16bf load_frag2(const __bf16* p0, const __bf16* p1) {
  v8bf lo = *(const v8bf*)p0;
  v8bf hi = *(const v8bf*)p1;
  v16bf r;
#pragma unroll
  for (int j = 0; j < 8; ++j) { r[j] = lo[j]; r[j + 8] = hi[j]; }
  return r;
}

__device__ __forceinline__ v8f wmma_bf16(v16bf a, v16bf b, v8f c) {
  return __builtin_amdgcn_wmma_f32_16x16x32_bf16(
      /*neg_a=*/false, a, /*neg_b=*/false, b,
      /*c_mod=*/(short)0, c, /*reuse_a=*/false, /*reuse_b=*/false);
}

// ---------------------------------------------------------------------------
// f32 -> bf16 elementwise convert
// ---------------------------------------------------------------------------
__global__ __launch_bounds__(256) void cvt_f32_bf16_kernel(const float* __restrict__ in,
                                                           __bf16* __restrict__ out,
                                                           int n) {
  int i = blockIdx.x * blockDim.x + threadIdx.x;
  int stride = gridDim.x * blockDim.x;
  for (; i < n; i += stride) out[i] = (__bf16)in[i];
}

// ---------------------------------------------------------------------------
// Transpose f32 [K][N] -> bf16 [N][K] (LDS tiled, 32x32 tiles, 256 threads)
// ---------------------------------------------------------------------------
__global__ __launch_bounds__(256) void transpose_cvt_kernel(const float* __restrict__ in,
                                                            __bf16* __restrict__ out,
                                                            int K, int N) {
  __shared__ float tile[32][33];
  int kb = blockIdx.x * 32;
  int nb = blockIdx.y * 32;
  int tx = threadIdx.x & 31;
  int ty = threadIdx.x >> 5;  // 0..7
#pragma unroll
  for (int r = ty; r < 32; r += 8)
    tile[r][tx] = in[(size_t)(kb + r) * N + (nb + tx)];
  __syncthreads();
#pragma unroll
  for (int r = ty; r < 32; r += 8)
    out[(size_t)(nb + r) * K + (kb + tx)] = (__bf16)tile[tx][r];
}

// ---------------------------------------------------------------------------
// bf16 GEMM: C[M][N] (f32) = A[M][K] (bf16, row-major) * Bt[N][K] (bf16) + bias[N]
// One wave computes a 32x64 output tile: 2 A-frags x 4 B-frags -> 8 WMMAs per
// 32-deep k-step. Ping-pong double buffer (unroll by 2) so buffers alternate
// roles with NO register copies; loads for one phase overlap the other phase's
// WMMAs. Requires K % 64 == 0.
// ---------------------------------------------------------------------------
__device__ __forceinline__ void gemm_load_step(const __bf16* __restrict__ arow0,
                                               const __bf16* __restrict__ arow1,
                                               const __bf16* const* __restrict__ brow,
                                               int k, v16bf af[2], v16bf bf[4]) {
  af[0] = load_frag2(arow0 + k, arow0 + k + 16);
  af[1] = load_frag2(arow1 + k, arow1 + k + 16);
#pragma unroll
  for (int j = 0; j < 4; ++j) bf[j] = load_frag2(brow[j] + k, brow[j] + k + 16);
}

__device__ __forceinline__ void gemm_compute_step(const v16bf af[2], const v16bf bf[4],
                                                  v8f acc[2][4]) {
#pragma unroll
  for (int mi = 0; mi < 2; ++mi)
#pragma unroll
    for (int j = 0; j < 4; ++j)
      acc[mi][j] = wmma_bf16(af[mi], bf[j], acc[mi][j]);
}

__global__ __launch_bounds__(256) void gemm_bf16_kernel(const __bf16* __restrict__ A,
                                                        const __bf16* __restrict__ Bt,
                                                        const float* __restrict__ bias,
                                                        float* __restrict__ C,
                                                        int M, int N, int K) {
  const int wave = blockIdx.x * 8 + (threadIdx.x >> 5);
  const int lane = threadIdx.x & 31;
  const int mtiles = M >> 5;   // 32-row tiles
  const int ntiles = N >> 6;   // 64-col tiles
  if (wave >= mtiles * ntiles) return;
  const int nt = wave / mtiles;
  const int mt = wave % mtiles;
  const int g = lane >> 4;
  const int n = lane & 15;

  v8f acc[2][4];
#pragma unroll
  for (int mi = 0; mi < 2; ++mi)
#pragma unroll
    for (int j = 0; j < 4; ++j)
#pragma unroll
      for (int r = 0; r < 8; ++r) acc[mi][j][r] = 0.f;

  const __bf16* arow0 = A + (size_t)(mt * 32 + n) * K + g * 8;        // M = mt*32 + n
  const __bf16* arow1 = A + (size_t)(mt * 32 + 16 + n) * K + g * 8;   // M = mt*32 + 16 + n
  const __bf16* brow[4];
#pragma unroll
  for (int j = 0; j < 4; ++j)
    brow[j] = Bt + (size_t)(nt * 64 + j * 16 + n) * K + g * 8;

  v16bf afA[2], bfA[4], afB[2], bfB[4];
  gemm_load_step(arow0, arow1, brow, 0, afA, bfA);

  for (int k0 = 0; k0 < K; k0 += 64) {
    // Phase B loads (k0+32) issued before phase A computes.
    gemm_load_step(arow0, arow1, brow, k0 + 32, afB, bfB);
    if (k0 + 64 < K) {
      __builtin_prefetch(arow0 + k0 + 64, 0, 3);
      __builtin_prefetch(arow1 + k0 + 64, 0, 3);
#pragma unroll
      for (int j = 0; j < 4; ++j) __builtin_prefetch(brow[j] + k0 + 64, 0, 3);
    }
    gemm_compute_step(afA, bfA, acc);
    // Phase A loads (k0+64) issued before phase B computes.
    if (k0 + 64 < K) gemm_load_step(arow0, arow1, brow, k0 + 64, afA, bfA);
    gemm_compute_step(afB, bfB, acc);
  }

#pragma unroll
  for (int mi = 0; mi < 2; ++mi)
#pragma unroll
    for (int j = 0; j < 4; ++j) {
      const int col = nt * 64 + j * 16 + n;
      const float b = bias ? bias[col] : 0.f;
#pragma unroll
      for (int r = 0; r < 8; ++r) {
        const int row = mt * 32 + mi * 16 + 8 * g + r;  // C layout: VGPR r -> M = 8*g + r
        C[(size_t)row * N + col] = acc[mi][j][r] + b;
      }
    }
}

// ---------------------------------------------------------------------------
// RoPE + split: qkv f32 [T][3][16][128] -> Q,K bf16 [T][2048], V bf16 [16][128][T]
// ---------------------------------------------------------------------------
__global__ __launch_bounds__(256) void rope_split_kernel(const float* __restrict__ qkv,
                                                         const float* __restrict__ cosb,
                                                         const float* __restrict__ sinb,
                                                         __bf16* __restrict__ Q,
                                                         __bf16* __restrict__ Kc,
                                                         __bf16* __restrict__ Vt) {
  const int t = blockIdx.x;
  const float* base = qkv + (size_t)t * 3 * HID;
  // RoPE on q and k: 1024 (head, d<64) pairs each
  for (int i = threadIdx.x; i < 1024; i += 256) {
    const int h = i >> 6;
    const int d = i & 63;
    const float c = cosb[t * 64 + d];
    const float s = sinb[t * 64 + d];
    {
      const float x1 = base[h * 128 + d];
      const float x2 = base[h * 128 + 64 + d];
      Q[(size_t)t * HID + h * 128 + d]      = (__bf16)(x1 * c - x2 * s);
      Q[(size_t)t * HID + h * 128 + 64 + d] = (__bf16)(x2 * c + x1 * s);
    }
    {
      const float x1 = base[HID + h * 128 + d];
      const float x2 = base[HID + h * 128 + 64 + d];
      Kc[(size_t)t * HID + h * 128 + d]      = (__bf16)(x1 * c - x2 * s);
      Kc[(size_t)t * HID + h * 128 + 64 + d] = (__bf16)(x2 * c + x1 * s);
    }
  }
  // V transposed: Vt[(h*128+d)][t]
  for (int i = threadIdx.x; i < 2048; i += 256) {
    Vt[(size_t)i * T_TOTAL + t] = (__bf16)base[2 * HID + i];
  }
}

// ---------------------------------------------------------------------------
// Flash attention. One wave = one (head, 16-row q tile).
//   S^T = K * Q^T   (A = K tile 16keys x 32d, B = Q^T, C = [key][q])
//   online softmax per q-column (one lane-half shuffle for the reduction)
//   O^T = V^T * P^T (A = Vt tile 16d x 32keys, B = P^T built via half-wave shfl)
// Output written row-major bf16 [t][h*128+d] for the output GEMM.
// ---------------------------------------------------------------------------
__global__ __launch_bounds__(128) void attn_fa_kernel(const __bf16* __restrict__ Q,
                                                      const __bf16* __restrict__ Kc,
                                                      const __bf16* __restrict__ Vt,
                                                      const int* __restrict__ cu, int ncu,
                                                      __bf16* __restrict__ O) {
  const int wave = blockIdx.x * 4 + (threadIdx.x >> 5);
  const int lane = threadIdx.x & 31;
  const int h  = wave >> 7;   // 2048/16 = 128 q-tiles per head
  const int qt = wave & 127;
  const int q0 = qt << 4;
  const int g = lane >> 4;
  const int n = lane & 15;
  const int qcol = q0 + n;

  int seg = 0;
  for (int i = 0; i < ncu - 1; ++i) { int c = cu[i]; if (c <= q0) seg = c; }

  // Q as B operand: lane holds column q = qcol, 8 d at (dc*32 + g*8) and 8 at (dc*32 + 16 + g*8)
  v16bf qf[4];
#pragma unroll
  for (int dc = 0; dc < 4; ++dc) {
    const __bf16* qp = Q + (size_t)qcol * HID + h * HSIZE + dc * 32 + g * 8;
    qf[dc] = load_frag2(qp, qp + 16);
  }

  v8f oacc[8];
#pragma unroll
  for (int dt = 0; dt < 8; ++dt)
#pragma unroll
    for (int r = 0; r < 8; ++r) oacc[dt][r] = 0.f;

  float mrun = -3.0e38f;
  float srun = 0.f;
  const float scale = 0.08838834764831845f;  // 128^-0.5
  const int kend = q0 + 16;

  for (int kt = seg; kt < kend; kt += 32) {
    // Prefetch next key tile's K rows while we compute on this one.
    if (kt + 63 < T_TOTAL) {
      __builtin_prefetch(Kc + (size_t)(kt + 32 + n) * HID + h * HSIZE, 0, 3);
      __builtin_prefetch(Kc + (size_t)(kt + 48 + n) * HID + h * HSIZE, 0, 3);
    }
    v8f s0, s1;
#pragma unroll
    for (int r = 0; r < 8; ++r) { s0[r] = 0.f; s1[r] = 0.f; }
#pragma unroll
    for (int dc = 0; dc < 4; ++dc) {
      const __bf16* kp0 = Kc + (size_t)(kt + n) * HID + h * HSIZE + dc * 32 + g * 8;
      v16bf ka0 = load_frag2(kp0, kp0 + 16);
      s0 = wmma_bf16(ka0, qf[dc], s0);
      const __bf16* kp1 = Kc + (size_t)(kt + 16 + n) * HID + h * HSIZE + dc * 32 + g * 8;
      v16bf ka1 = load_frag2(kp1, kp1 + 16);
      s1 = wmma_bf16(ka1, qf[dc], s1);
    }
    // scale + causal mask (C layout: VGPR r -> key row 8*g + r within tile)
    float v0[8], v1[8];
#pragma unroll
    for (int r = 0; r < 8; ++r) {
      const int k0i = kt + 8 * g + r;
      const int k1i = k0i + 16;
      v0[r] = (k0i <= qcol) ? s0[r] * scale : -3.0e38f;
      v1[r] = (k1i <= qcol) ? s1[r] * scale : -3.0e38f;
    }
    float mt = -3.0e38f;
#pragma unroll
    for (int r = 0; r < 8; ++r) { mt = fmaxf(mt, v0[r]); mt = fmaxf(mt, v1[r]); }
    mt = fmaxf(mt, __shfl_xor(mt, 16, 32));
    const float mnew = fmaxf(mrun, mt);
    const float alpha = __expf(mrun - mnew);
    float e0[8], e1[8];
    float lsum = 0.f;
#pragma unroll
    for (int r = 0; r < 8; ++r) {
      e0[r] = __expf(v0[r] - mnew);
      e1[r] = __expf(v1[r] - mnew);
      lsum += e0[r] + e1[r];
    }
    srun = srun * alpha + lsum;
    mrun = mnew;
#pragma unroll
    for (int dt = 0; dt < 8; ++dt)
#pragma unroll
      for (int r = 0; r < 8; ++r) oacc[dt][r] *= alpha;

    // Build P^T B-fragment (32 keys x 16 q) from the two C-layout score tiles:
    //   lane-half 0 needs keys kt..kt+15 (tile0 rows 0..7 own, 8..15 from lane^16)
    //   lane-half 1 needs keys kt+16..kt+31 (tile1 rows 0..7 from lane^16, 8..15 own)
    v16bf pf;
#pragma unroll
    for (int j = 0; j < 8; ++j) {
      const float x0 = __shfl_xor(e0[j], 16, 32);
      const float x1 = __shfl_xor(e1[j], 16, 32);
      pf[j]     = (__bf16)(g ? x1 : e0[j]);
      pf[j + 8] = (__bf16)(g ? e1[j] : x0);
    }
#pragma unroll
    for (int dt = 0; dt < 8; ++dt) {
      const __bf16* vp = Vt + (size_t)(h * HSIZE + dt * 16 + n) * T_TOTAL + kt + g * 16;
      v16bf va = load_frag2(vp, vp + 8);
      oacc[dt] = wmma_bf16(va, pf, oacc[dt]);
      if (kt + 63 < T_TOTAL) __builtin_prefetch(vp + 32, 0, 3);  // next key tile, same row
    }
  }

  const float stot = srun + __shfl_xor(srun, 16, 32);
  const float rinv = 1.0f / stot;
#pragma unroll
  for (int dt = 0; dt < 8; ++dt) {
#pragma unroll
    for (int r = 0; r < 8; ++r) {
      const int d = dt * 16 + 8 * g + r;  // O^T C layout: row = d, col = q
      O[(size_t)qcol * HID + h * HSIZE + d] = (__bf16)(oacc[dt][r] * rinv);
    }
  }
}

// ---------------------------------------------------------------------------
// Host-side orchestration
// ---------------------------------------------------------------------------
extern "C" void kernel_launch(void* const* d_in, const int* in_sizes, int n_in,
                              void* d_out, int out_size, void* d_ws, size_t ws_size,
                              hipStream_t stream) {
  const float* hidden = (const float*)d_in[0];   // [2048][2048]
  const float* cosb   = (const float*)d_in[1];   // [2048][64]
  const float* sinb   = (const float*)d_in[2];   // [2048][64]
  const float* W_qkv  = (const float*)d_in[3];   // [2048][6144]
  const float* b_qkv  = (const float*)d_in[4];   // [6144]
  const float* W_o    = (const float*)d_in[5];   // [2048][2048]
  const float* b_o    = (const float*)d_in[6];   // [2048]
  const int*   cu     = (const int*)d_in[7];     // [5]
  const int    ncu    = in_sizes[7];
  float* out = (float*)d_out;

  // Workspace layout (256-byte aligned slices)
  char* ws = (char*)d_ws;
  size_t off = 0;
  auto take = [&](size_t bytes) {
    size_t o = off;
    off = (off + bytes + 255) & ~(size_t)255;
    return o;
  };
  const size_t o_hsb   = take((size_t)T_TOTAL * HID * 2);        // hidden bf16
  const size_t o_wqkvt = take((size_t)3 * HID * HID * 2);        // W_qkv^T bf16 [6144][2048]
  const size_t o_wot   = take((size_t)HID * HID * 2);            // W_o^T bf16 [2048][2048]
  const size_t o_qkvf  = take((size_t)T_TOTAL * 3 * HID * 4);    // qkv f32
  const size_t o_qb    = take((size_t)T_TOTAL * HID * 2);        // Q bf16
  const size_t o_kb    = take((size_t)T_TOTAL * HID * 2);        // K bf16
  const size_t o_vtb   = take((size_t)T_TOTAL * HID * 2);        // V^T bf16 [h][d][t]
  const size_t o_attnb = take((size_t)T_TOTAL * HID * 2);        // attn bf16
  (void)ws_size;

  __bf16* hsb   = (__bf16*)(ws + o_hsb);
  __bf16* wqkvt = (__bf16*)(ws + o_wqkvt);
  __bf16* wot   = (__bf16*)(ws + o_wot);
  float*  qkvf  = (float*)(ws + o_qkvf);
  __bf16* qb    = (__bf16*)(ws + o_qb);
  __bf16* kb    = (__bf16*)(ws + o_kb);
  __bf16* vtb   = (__bf16*)(ws + o_vtb);
  __bf16* attnb = (__bf16*)(ws + o_attnb);

  // 1. hidden -> bf16
  cvt_f32_bf16_kernel<<<2048, 256, 0, stream>>>(hidden, hsb, T_TOTAL * HID);
  // 2. weight transposes (f32 [K][N] -> bf16 [N][K])
  transpose_cvt_kernel<<<dim3(HID / 32, 3 * HID / 32), 256, 0, stream>>>(W_qkv, wqkvt, HID, 3 * HID);
  transpose_cvt_kernel<<<dim3(HID / 32, HID / 32), 256, 0, stream>>>(W_o, wot, HID, HID);
  // 3. QKV GEMM: [2048][2048] x [2048][6144] -> f32 qkv
  {
    const int waves = (T_TOTAL / 32) * (3 * HID / 64);  // 64 * 96 = 6144
    gemm_bf16_kernel<<<waves / 8, 256, 0, stream>>>(hsb, wqkvt, b_qkv, qkvf,
                                                    T_TOTAL, 3 * HID, HID);
  }
  // 4. RoPE + split
  rope_split_kernel<<<T_TOTAL, 256, 0, stream>>>(qkvf, cosb, sinb, qb, kb, vtb);
  // 5. Flash attention: 16 heads x 128 q-tiles = 2048 waves, 4 waves/block
  attn_fa_kernel<<<512, 128, 0, stream>>>(qb, kb, vtb, cu, ncu, attnb);
  // 6. Output GEMM: [2048][2048] x [2048][2048] + b_o -> f32 out
  {
    const int waves = (T_TOTAL / 32) * (HID / 64);  // 64 * 32 = 2048
    gemm_bf16_kernel<<<waves / 8, 256, 0, stream>>>(attnb, wot, b_o, out,
                                                    T_TOTAL, HID, HID);
  }
}
